// TopKRouter_37589553774753
// MI455X (gfx1250) — matrix operand order repaired and checked by reading, not compiled, not run on hardware
//
#include <hip/hip_runtime.h>
#include <hip/hip_bf16.h>

typedef __attribute__((ext_vector_type(16))) __bf16 v16bf;
typedef __attribute__((ext_vector_type(8)))  __bf16 v8bf;
typedef __attribute__((ext_vector_type(8)))  float  v8f;
typedef __attribute__((ext_vector_type(4)))  float  v4f;

#define TOKENS 16384
#define DMODEL 2048
#define NEXP   64
#define KSLAB  256                 // k-values staged in LDS per slab
#define WROW   (KSLAB + 8)         // padded bf16 row stride (528B: 16B aligned, conflict-friendly)
#define SROW   68                  // padded score row stride in floats (bank-conflict-free scan)

#define LDS_W_BYTES   (NEXP * WROW * 2)      // 33792
#define LDS_S_BYTES   (8 * 16 * SROW * 4)    // 34816
#define LDS_BYTES     (LDS_S_BYTES > LDS_W_BYTES ? LDS_S_BYTES : LDS_W_BYTES)

#if defined(__has_builtin)
#  if __has_builtin(__builtin_amdgcn_sched_barrier)
#    define SCHED_FENCE() __builtin_amdgcn_sched_barrier(0)
#  endif
#endif
#ifndef SCHED_FENCE
#  define SCHED_FENCE()
#endif

__global__ __launch_bounds__(256)
void topk_router_kernel(const float* __restrict__ x,
                        const float* __restrict__ W,
                        int*   __restrict__ out_idx,
                        float* __restrict__ out_w)
{
    __shared__ __align__(16) unsigned char smem_raw[LDS_BYTES];
    __bf16* ldsW = (__bf16*)smem_raw;
    float*  ldsS = (float*)smem_raw;

    const int tid  = threadIdx.x;
    const int wave = tid >> 5;
    const int lane = tid & 31;
    const int ln16 = lane & 15;
    const int hi   = lane >> 4;            // 0: K 0-7 and 16-23, 1: K 8-15 and 24-31

    // one 16-token tile per wave; 128 blocks * 8 waves = 1024 tiles = 16384 tokens
    const int tile = blockIdx.x * 8 + wave;
    const int arow = tile * 16 + ln16;     // token row this lane feeds into the A fragment
    const float* xrow = x + (size_t)arow * DMODEL;

    v8f acc[4];
#pragma unroll
    for (int t = 0; t < 4; ++t)
        acc[t] = (v8f){0.f, 0.f, 0.f, 0.f, 0.f, 0.f, 0.f, 0.f};

    for (int ks = 0; ks < DMODEL; ks += KSLAB) {
        __syncthreads();
        // ---- cooperative stage of W[:, ks:ks+256] into LDS as bf16 ----
        // 4 threads per expert row, 64 contiguous floats each (16x float4)
        {
            const int e   = tid >> 2;
            const int seg = (tid & 3) * 64;
            const float* wp = W + (size_t)e * DMODEL + ks + seg;
            __bf16* lp = ldsW + e * WROW + seg;
#pragma unroll
            for (int j = 0; j < 16; ++j) {
                v4f f = *(const v4f*)(wp + j * 4);
                lp[j * 4 + 0] = (__bf16)f[0];
                lp[j * 4 + 1] = (__bf16)f[1];
                lp[j * 4 + 2] = (__bf16)f[2];
                lp[j * 4 + 3] = (__bf16)f[3];
            }
        }
        __syncthreads();

        // Prefetch this lane's ENTIRE next x slab (1KB/lane = 8 cachelines @128B)
        // so the HBM stream runs a full slab ahead of compute.
        if (ks + KSLAB < DMODEL) {
#pragma unroll
            for (int p = 0; p < 8; ++p)
                __builtin_prefetch(xrow + ks + KSLAB + p * 32, 0, 0);
        }

        // ---- 8 WMMA k-steps over this slab ----
#pragma unroll
        for (int kk = 0; kk < KSLAB; kk += 32) {
            // A fragment: 16x32 bf16. Lane holds K chunks [hi*8, +8) and [hi*8+16, +8)
            const float* ap = xrow + ks + kk + hi * 8;
            v4f f0 = *(const v4f*)(ap);
            v4f f1 = *(const v4f*)(ap + 4);
            v4f f2 = *(const v4f*)(ap + 16);
            v4f f3 = *(const v4f*)(ap + 20);
            v16bf a;
#pragma unroll
            for (int j = 0; j < 4; ++j) {
                a[j]      = (__bf16)f0[j];
                a[4 + j]  = (__bf16)f1[j];
                a[8 + j]  = (__bf16)f2[j];
                a[12 + j] = (__bf16)f3[j];
            }

            // Load ALL four B fragments first; the sched barrier stops the
            // compiler from sinking each ds_load pair down to its WMMA, so we
            // get one ds clause + one wait + four back-to-back WMMAs.
            v16bf b[4];
#pragma unroll
            for (int t = 0; t < 4; ++t) {
                // B fragment: 32x16 bf16, column n = ln16 (expert t*16+ln16),
                // lane holds K chunks [hi*8, +8) and [hi*8+16, +8)
                const __bf16* pb = ldsW + (t * 16 + ln16) * WROW + kk + hi * 8;
                v8bf b0 = *(const v8bf*)(pb);
                v8bf b1 = *(const v8bf*)(pb + 16);
#pragma unroll
                for (int j = 0; j < 8; ++j) { b[t][j] = b0[j]; b[t][8 + j] = b1[j]; }
            }
            SCHED_FENCE();
#pragma unroll
            for (int t = 0; t < 4; ++t) {
                acc[t] = __builtin_amdgcn_wmma_f32_16x16x32_bf16(
                    false, a, false, b[t], (short)0, acc[t], false, false);
            }
        }
    }

    // ---- scores -> LDS (per-wave 16x64 region, padded rows) ----
    __syncthreads();   // all waves done with the shared W slab buffer
    float* sc = ldsS + wave * 16 * SROW;
#pragma unroll
    for (int t = 0; t < 4; ++t) {
#pragma unroll
        for (int i = 0; i < 8; ++i) {
            const int m = i + hi * 8;               // C/D layout: vgpr i, hi half -> row i+8
            sc[m * SROW + t * 16 + ln16] = acc[t][i];
        }
    }
    __syncthreads();

    // ---- top-2 + softmax: lanes 0..15 each own one token ----
    if (lane < 16) {
        const float* s = sc + lane * SROW;
        float v0 = -3.4e38f, v1 = -3.4e38f;
        int   i0 = 0,        i1 = 0;
#pragma unroll 8
        for (int e = 0; e < NEXP; ++e) {
            float v = s[e];
            if (v > v0)      { v1 = v0; i1 = i0; v0 = v; i0 = e; }
            else if (v > v1) { v1 = v;  i1 = e; }
        }
        const int token = tile * 16 + lane;
        const float e1 = __expf(v1 - v0);
        const float rd = 1.f / (1.f + e1);
        out_idx[token * 2 + 0] = i0;
        out_idx[token * 2 + 1] = i1;
        out_w[token * 2 + 0] = rd;
        out_w[token * 2 + 1] = e1 * rd;
    }
}

extern "C" void kernel_launch(void* const* d_in, const int* in_sizes, int n_in,
                              void* d_out, int out_size, void* d_ws, size_t ws_size,
                              hipStream_t stream) {
    const float* x = (const float*)d_in[0];   // [16384, 2048] fp32
    const float* W = (const float*)d_in[1];   // [64, 2048] fp32
    // d_out: [16384*2] int32 indices, then [16384*2] fp32 weights (flat concat)
    int*   out_idx = (int*)d_out;
    float* out_w   = (float*)d_out + (size_t)TOKENS * 2;

    topk_router_kernel<<<128, 256, 0, stream>>>(x, W, out_idx, out_w);
}
